// DeepDFTeqv_Model_2619930050965
// MI455X (gfx1250) — matrix operand order repaired and checked by reading, not compiled, not run on hardware
//
#include <hip/hip_runtime.h>

#define NF 32
#define NK 20
#define NA 384
#define NPR 1024
#define CUTOFF 15.0f
#define BOHR 0.5291772105638411f
#define EPSX 1e-11f
#define PI_OVER_CUT 0.2094395102393195f   // pi / 15

typedef __attribute__((ext_vector_type(16))) _Float16 v16h;
typedef __attribute__((ext_vector_type(8)))  float    v8f;

__device__ __forceinline__ float silu_f(float x) { return x / (1.f + __expf(-x)); }

// ---------------------------------------------------------------- init
__global__ void init_kernel(const int* __restrict__ anum, const float* __restrict__ embed,
                            float* __restrict__ sA, float* __restrict__ vA,
                            float* __restrict__ sP, float* __restrict__ vP) {
  int idx = blockIdx.x * blockDim.x + threadIdx.x;
  if (idx < NA * NF)      sA[idx] = embed[anum[idx / NF] * NF + (idx & (NF - 1))];
  if (idx < NA * 3 * NF)  vA[idx] = 0.f;
  if (idx < NPR * NF)     sP[idx] = 0.f;
  if (idx < NPR * 3 * NF) vP[idx] = 0.f;
}

// ----------------------------------------------- per-source "left" MLP
// left[s, 0:96] = silu(s @ w1 + b1) @ w2 + b2   (one wave per node, lane = feature)
__global__ void left_kernel(const float* __restrict__ sfeat, int nnode,
                            const float* __restrict__ w1, const float* __restrict__ b1,
                            const float* __restrict__ w2, const float* __restrict__ b2,
                            float* __restrict__ left) {
  int lane = threadIdx.x & 31;
  int node = blockIdx.x * (blockDim.x >> 5) + (threadIdx.x >> 5);
  if (node >= nnode) return;
  float sf = sfeat[node * NF + lane];
  float h = b1[lane];
  #pragma unroll
  for (int i = 0; i < NF; ++i) h += __shfl(sf, i) * w1[i * NF + lane];
  h = silu_f(h);
  #pragma unroll
  for (int p = 0; p < 3; ++p) {
    int ch = p * NF + lane;
    float acc = b2[ch];
    #pragma unroll
    for (int g = 0; g < NF; ++g) acc += __shfl(h, g) * w2[g * 96 + ch];
    left[node * 96 + ch] = acc;
  }
}

// ------------------------------------------------------- message kernel
// One wave per target t. For each 16-source tile: WMMA computes the
// (16 x 20) @ (20 x 16) tiles of right = expansion @ wr, with the pre-swizzled
// f16 B fragments staged in LDS (shared by all 8 waves) and consumed one tile
// at a time to keep register pressure low. Per-row geometry is computed once
// per lane (row = lane&15) and redistributed with shuffles.
__global__ __launch_bounds__(256) void msg_kernel(
    const float* __restrict__ tgt, int ntgt,
    const float* __restrict__ src, int nsrc,
    const float* __restrict__ left,
    const float* __restrict__ vsrc,
    const float* __restrict__ wr, const float* __restrict__ br,
    float* __restrict__ dS, float* __restrict__ dV) {
  __shared__ __align__(32) _Float16 ldsB[6 * 32 * 16];   // 6 KB: B fragments per lane
  __shared__ float ldsC[NA * 3];                          // source coords

  // cooperative fill: ldsB[nt*512 + lane*16 + i] = wr[k, ch], k = (lane>>4)*16 + i
  for (int e = threadIdx.x; e < 6 * 32 * 16; e += blockDim.x) {
    int nt = e >> 9;
    int lane_e = (e >> 4) & 31;
    int i = e & 15;
    int k = ((lane_e >> 4) << 4) + i;
    int ch = nt * 16 + (lane_e & 15);
    ldsB[e] = (k < NK) ? (_Float16)wr[k * 96 + ch] : (_Float16)0.f;
  }
  for (int e = threadIdx.x; e < nsrc * 3; e += blockDim.x) ldsC[e] = src[e];
  __syncthreads();

  int lane = threadIdx.x & 31;
  int t = blockIdx.x * (blockDim.x >> 5) + (threadIdx.x >> 5);
  if (t >= ntgt) return;                     // wave-uniform
  int n = lane & 15;
  int half = lane >> 4;

  float brv[6];
  #pragma unroll
  for (int nt = 0; nt < 6; ++nt) brv[nt] = br[nt * 16 + n];

  float tx = tgt[t * 3 + 0], ty = tgt[t * 3 + 1], tz = tgt[t * 3 + 2];
  float accS0 = 0.f, accS1 = 0.f;
  float accV[3][2] = {{0.f, 0.f}, {0.f, 0.f}, {0.f, 0.f}};

  for (int sb = 0; sb < nsrc; sb += 16) {
    // ---- geometry for this lane's row (lane&15), shared later via shuffle ----
    int sa = sb + n;
    float ax = tx - ldsC[sa * 3 + 0];
    float ay = ty - ldsC[sa * 3 + 1];
    float az = tz - ldsC[sa * 3 + 2];
    float da = sqrtf(fmaxf(ax * ax + ay * ay + az * az, 1e-20f));
    float inv = 1.f / da;
    float uxA = ax * inv, uyA = ay * inv, uzA = az * inv;
    float fcA = (da < CUTOFF && da > 1e-9f)
                  ? 0.5f * (__cosf(da * PI_OVER_CUT) + 1.f) : 0.f;  // cutoff * mask

    // ---- A fragment: sinc expansion of row (lane&15), 16x32 f16 A layout ----
    float xA = da * BOHR + EPSX;
    float rxA = 1.f / xA;
    v16h A;
    #pragma unroll
    for (int r = 0; r < 8; ++r) {
      int bk = 2 * r + half * 16;
      A[2 * r]     = (bk     < NK) ? (_Float16)(__sinf(xA * ((bk + 1) * PI_OVER_CUT)) * rxA)
                                   : (_Float16)0.f;
      A[2 * r + 1] = (bk + 1 < NK) ? (_Float16)(__sinf(xA * ((bk + 2) * PI_OVER_CUT)) * rxA)
                                   : (_Float16)0.f;
    }

    // ---- one WMMA tile at a time; consume D immediately ----
    v8f zero = {};
    #pragma unroll
    for (int p = 0; p < 3; ++p) {          // 0: a-split, 1: b-split, 2: c-split
      #pragma unroll
      for (int j = 0; j < 2; ++j) {        // low/high 16 channels of the split
        int nt = p * 2 + j;
        v16h Bf = *(const v16h*)&ldsB[nt * 512 + lane * 16];
        v8f D = __builtin_amdgcn_wmma_f32_16x16x32_f16(false, A, false, Bf,
                                                       (short)0, zero, false, false);
        #pragma unroll
        for (int r = 0; r < 8; ++r) {
          int row = r + 8 * half;          // source row this D element belongs to
          int s = sb + row;
          float fc = __shfl(fcA, row);
          float e = (D[r] + brv[nt]) * fc * left[s * 96 + p * 32 + j * 16 + n];
          if (p == 1) {
            if (j == 0) accS0 += e; else accS1 += e;
          } else if (p == 0) {
            const float* Vs = vsrc + s * 96;
            int f = j * 16 + n;
            accV[0][j] += e * Vs[f];
            accV[1][j] += e * Vs[32 + f];
            accV[2][j] += e * Vs[64 + f];
          } else {
            accV[0][j] += e * __shfl(uxA, row);
            accV[1][j] += e * __shfl(uyA, row);
            accV[2][j] += e * __shfl(uzA, row);
          }
        }
      }
    }
  }

  // fold upper-half rows (8..15) into lower-half lanes (same channel column)
  accS0 += __shfl_xor(accS0, 16);
  accS1 += __shfl_xor(accS1, 16);
  #pragma unroll
  for (int x = 0; x < 3; ++x) {
    accV[x][0] += __shfl_xor(accV[x][0], 16);
    accV[x][1] += __shfl_xor(accV[x][1], 16);
  }
  if (half == 0) {
    dS[t * 32 + n]      = accS0;
    dS[t * 32 + 16 + n] = accS1;
    #pragma unroll
    for (int x = 0; x < 3; ++x) {
      dV[t * 96 + x * 32 + n]      = accV[x][0];
      dV[t * 96 + x * 32 + 16 + n] = accV[x][1];
    }
  }
}

// ---------------------------------------------------------------- add deltas
__global__ void add_kernel(float* __restrict__ s, float* __restrict__ v,
                           const float* __restrict__ dS, const float* __restrict__ dV,
                           int nnode) {
  int idx = blockIdx.x * blockDim.x + threadIdx.x;
  if (idx < nnode * NF)     s[idx] += dS[idx];
  if (idx < nnode * 3 * NF) v[idx] += dV[idx];
}

// ---------------------------------------------------------------- update
// One wave per node, lane = feature channel (wave32 == NF).
__global__ void update_kernel(float* __restrict__ sfeat, float* __restrict__ vfeat, int nnode,
                              const float* __restrict__ U, const float* __restrict__ V,
                              const float* __restrict__ w1, const float* __restrict__ b1,
                              const float* __restrict__ w2, const float* __restrict__ b2) {
  int lane = threadIdx.x & 31;
  int node = blockIdx.x * (blockDim.x >> 5) + (threadIdx.x >> 5);
  if (node >= nnode) return;
  float sf = sfeat[node * NF + lane];
  float v0 = vfeat[node * 96 + lane];
  float v1 = vfeat[node * 96 + NF + lane];
  float v2 = vfeat[node * 96 + 2 * NF + lane];
  float l0 = 0.f, l1 = 0.f, l2 = 0.f, r0 = 0.f, r1 = 0.f, r2 = 0.f;
  #pragma unroll
  for (int g = 0; g < NF; ++g) {
    float uu = U[g * NF + lane], vv = V[g * NF + lane];
    float a0 = __shfl(v0, g), a1 = __shfl(v1, g), a2 = __shfl(v2, g);
    l0 += a0 * uu; l1 += a1 * uu; l2 += a2 * uu;
    r0 += a0 * vv; r1 += a1 * vv; r2 += a2 * vv;
  }
  float rn = sqrtf(r0 * r0 + r1 * r1 + r2 * r2);
  float h = b1[lane];
  #pragma unroll
  for (int i = 0; i < NF; ++i) h += __shfl(sf, i) * w1[i * NF + lane];
  #pragma unroll
  for (int i = 0; i < NF; ++i) h += __shfl(rn, i) * w1[(NF + i) * NF + lane];
  h = silu_f(h);
  float fa = b2[lane], fb = b2[NF + lane], fcc = b2[2 * NF + lane];
  #pragma unroll
  for (int g = 0; g < NF; ++g) {
    float hg = __shfl(h, g);
    fa  += hg * w2[g * 96 + lane];
    fb  += hg * w2[g * 96 + NF + lane];
    fcc += hg * w2[g * 96 + 2 * NF + lane];
  }
  float dot = l0 * r0 + l1 * r1 + l2 * r2;
  sfeat[node * NF + lane] = sf + dot * fb + fcc;
  vfeat[node * 96 + lane]          = v0 + fa * l0;
  vfeat[node * 96 + NF + lane]     = v1 + fa * l1;
  vfeat[node * 96 + 2 * NF + lane] = v2 + fa * l2;
}

// ---------------------------------------------------------------- output
__global__ void out_kernel(const float* __restrict__ sP, const float* __restrict__ vP,
                           float* __restrict__ out) {
  int idx = blockIdx.x * blockDim.x + threadIdx.x;
  if (idx < NPR * NF) out[idx] = sP[idx];
  int idx2 = idx - NPR * NF;
  if (idx2 >= 0 && idx2 < NPR * 3 * NF) out[NPR * NF + idx2] = vP[idx2];
}

extern "C" void kernel_launch(void* const* d_in, const int* in_sizes, int n_in,
                              void* d_out, int out_size, void* d_ws, size_t ws_size,
                              hipStream_t stream) {
  (void)in_sizes; (void)n_in; (void)out_size; (void)ws_size;
  const float* atom_coords  = (const float*)d_in[0];
  const float* probe_coords = (const float*)d_in[1];
  const int*   atom_numbers = (const int*)d_in[2];
  // params, sorted-key order
  const float* embed   = (const float*)d_in[3];
  const float* msg_b1  = (const float*)d_in[4];
  const float* msg_b2  = (const float*)d_in[5];
  const float* msg_br  = (const float*)d_in[6];
  const float* msg_w1  = (const float*)d_in[7];
  const float* msg_w2  = (const float*)d_in[8];
  const float* msg_wr  = (const float*)d_in[9];
  const float* msgp_b1 = (const float*)d_in[10];
  const float* msgp_b2 = (const float*)d_in[11];
  const float* msgp_br = (const float*)d_in[12];
  const float* msgp_w1 = (const float*)d_in[13];
  const float* msgp_w2 = (const float*)d_in[14];
  const float* msgp_wr = (const float*)d_in[15];
  const float* upd_U   = (const float*)d_in[16];
  const float* upd_V   = (const float*)d_in[17];
  const float* upd_b1  = (const float*)d_in[18];
  const float* upd_b2  = (const float*)d_in[19];
  const float* upd_w1  = (const float*)d_in[20];
  const float* upd_w2  = (const float*)d_in[21];
  const float* updp_U  = (const float*)d_in[22];
  const float* updp_V  = (const float*)d_in[23];
  const float* updp_b1 = (const float*)d_in[24];
  const float* updp_b2 = (const float*)d_in[25];
  const float* updp_w1 = (const float*)d_in[26];
  const float* updp_w2 = (const float*)d_in[27];

  float* ws  = (float*)d_ws;
  float* sA  = ws;                 // 384*32
  float* vA  = sA  + NA * NF;      // 384*96
  float* sP  = vA  + NA * 3 * NF;  // 1024*32
  float* vP  = sP  + NPR * NF;     // 1024*96
  float* lft = vP  + NPR * 3 * NF; // 384*96
  float* dS  = lft + NA * 96;      // 1024*32
  float* dV  = dS  + NPR * NF;     // 1024*96

  const int BLK = 256;             // 8 waves (wave32) per block
  init_kernel<<<(NPR * 3 * NF + BLK - 1) / BLK, BLK, 0, stream>>>(
      atom_numbers, embed, sA, vA, sP, vP);

  for (int d = 0; d < 3; ++d) {
    // ---- atom-atom message ----
    left_kernel<<<NA / 8, BLK, 0, stream>>>(
        sA, NA, msg_w1 + d * 1024, msg_b1 + d * 32, msg_w2 + d * 3072, msg_b2 + d * 96, lft);
    msg_kernel<<<NA / 8, BLK, 0, stream>>>(
        atom_coords, NA, atom_coords, NA, lft, vA,
        msg_wr + d * 1920, msg_br + d * 96, dS, dV);
    add_kernel<<<(NA * 3 * NF + BLK - 1) / BLK, BLK, 0, stream>>>(sA, vA, dS, dV, NA);
    update_kernel<<<NA / 8, BLK, 0, stream>>>(
        sA, vA, NA, upd_U + d * 1024, upd_V + d * 1024,
        upd_w1 + d * 2048, upd_b1 + d * 32, upd_w2 + d * 3072, upd_b2 + d * 96);

    // ---- atom-probe message ----
    left_kernel<<<NA / 8, BLK, 0, stream>>>(
        sA, NA, msgp_w1 + d * 1024, msgp_b1 + d * 32, msgp_w2 + d * 3072, msgp_b2 + d * 96, lft);
    msg_kernel<<<NPR / 8, BLK, 0, stream>>>(
        probe_coords, NPR, atom_coords, NA, lft, vA,
        msgp_wr + d * 1920, msgp_br + d * 96, dS, dV);
    add_kernel<<<(NPR * 3 * NF + BLK - 1) / BLK, BLK, 0, stream>>>(sP, vP, dS, dV, NPR);
    update_kernel<<<NPR / 8, BLK, 0, stream>>>(
        sP, vP, NPR, updp_U + d * 1024, updp_V + d * 1024,
        updp_w1 + d * 2048, updp_b1 + d * 32, updp_w2 + d * 3072, updp_b2 + d * 96);
  }

  out_kernel<<<(NPR * NF * 4 + BLK - 1) / BLK, BLK, 0, stream>>>(sP, vP, (float*)d_out);
}
